// KnowformerVLayer_15951508537886
// MI455X (gfx1250) — matrix operand order repaired and checked by reading, not compiled, not run on hardware
//
#include <hip/hip_runtime.h>
#include <hip/hip_bf16.h>

// ---------------------------------------------------------------------------
// KnowformerVLayer on MI455X (gfx1250, wave32)
//   B=4, V=20000, D=64, R=64, E=640000
// Phases:
//   0) zero agg scratch (V x 256 f32, 20.5 MB -> L2 resident)
//   1) relation = z @ Wz.T + bz  -> (R=64, B*D=256) f32, 64 KB
//   2) edge scatter: agg[dst] += relation[etype] * node_in[src]
//        relation staged in LDS, x gathered via L2, relaxed agent-scope
//        f32 atomics (L2-side RMW, no return)
//   3) per 16-node tile: h0 = agg^T + beta*x; two 64x64 GEMMs via
//        v_wmma_f32_16x16x32_f16 (f16 in, f32 accum); ReLU; LayerNorm;
//        out = h + x
// ---------------------------------------------------------------------------

#define BB 4
#define VV 20000
#define DD 64
#define RR 64
#define EE 640000
#define BD 256            // B*D
#define LN_EPS 1e-5f

typedef __attribute__((ext_vector_type(16))) _Float16 v16h;
typedef __attribute__((ext_vector_type(8)))  float    v8f;

// ---------------------------------------------------------------- zero agg --
__global__ __launch_bounds__(256)
void k_zero(float* __restrict__ p, int n) {
    int i = blockIdx.x * 256 + threadIdx.x;
    int stride = gridDim.x * 256;
    for (; i < n; i += stride) p[i] = 0.0f;
}

// ---------------------------------------------------- relation = z@Wz.T+bz --
// rel laid out as rel[r*256 + b*64 + d]
__global__ __launch_bounds__(256)
void k_relation(const float* __restrict__ z, const float* __restrict__ Wz,
                const float* __restrict__ bz, float* __restrict__ rel) {
    int tid = blockIdx.x * 256 + threadIdx.x;      // 0 .. 16383
    int r  = tid >> 8;
    int bd = tid & 255;
    int b  = bd >> 6;
    int d  = bd & 63;
    const float* zrow = z  + b * DD;
    const float* wrow = Wz + (size_t)(r * DD + d) * DD;
    float acc = bz[r * DD + d];
#pragma unroll
    for (int k = 0; k < DD; k += 4) {
        float4 zv = *(const float4*)(zrow + k);
        float4 wv = *(const float4*)(wrow + k);
        acc += zv.x * wv.x + zv.y * wv.y + zv.z * wv.z + zv.w * wv.w;
    }
    rel[r * BD + b * DD + d] = acc;
}

// -------------------------------------------------------------- edge phase --
// 256 threads = 4 groups of 64; each group handles one edge's 256 columns
// (64 lanes x float4). relation table (64 KB) staged in LDS per block.
__global__ __launch_bounds__(256)
void k_edges(const int* __restrict__ ei, const float* __restrict__ x,
             const float* __restrict__ rel, float* __restrict__ agg) {
    __shared__ __align__(16) float lrel[RR * BD];       // 64 KB
    for (int i = threadIdx.x; i < RR * BD; i += 256) lrel[i] = rel[i];
    __syncthreads();

    const int grp    = (blockIdx.x << 2) + (threadIdx.x >> 6);
    const int stride = gridDim.x << 2;
    const int c      = threadIdx.x & 63;     // column group: col = 4*c
    const int b      = c >> 4;
    const int d      = (c & 15) << 2;

    for (int e = grp; e < EE; e += stride) {
        const int src = ei[e * 3 + 0];
        const int et  = ei[e * 3 + 1];
        const int dst = ei[e * 3 + 2];
        if (e + stride < EE)                              // gfx1250 prefetch
            __builtin_prefetch(ei + (size_t)(e + stride) * 3, 0, 1);

        const float4 r4 = *(const float4*)(&lrel[et * BD + c * 4]);
        const float4 n4 = *(const float4*)(x + ((size_t)b * VV + src) * DD + d);
        float* ap = agg + (size_t)dst * BD + c * 4;
        __hip_atomic_fetch_add(ap + 0, r4.x * n4.x, __ATOMIC_RELAXED, __HIP_MEMORY_SCOPE_AGENT);
        __hip_atomic_fetch_add(ap + 1, r4.y * n4.y, __ATOMIC_RELAXED, __HIP_MEMORY_SCOPE_AGENT);
        __hip_atomic_fetch_add(ap + 2, r4.z * n4.z, __ATOMIC_RELAXED, __HIP_MEMORY_SCOPE_AGENT);
        __hip_atomic_fetch_add(ap + 3, r4.w * n4.w, __ATOMIC_RELAXED, __HIP_MEMORY_SCOPE_AGENT);
    }
}

// ------------------------------------------- MLP (WMMA) + LayerNorm + resid --
// 128 threads = 4 waves; wave w handles tile = blockIdx.x*4 + w
// tile -> (b, 16-row slab of v). A staged in LDS as f16; B fragments built
// per-lane straight from row-major W1/W2 (lane%16 = output col, K contiguous).
__global__ __launch_bounds__(128)
void k_mlp_ln(const float* __restrict__ agg, const float* __restrict__ x,
              const float* __restrict__ W1, const float* __restrict__ b1,
              const float* __restrict__ W2, const float* __restrict__ b2,
              const float* __restrict__ beta, const float* __restrict__ lnw,
              const float* __restrict__ lnb, float* __restrict__ out) {
    __shared__ __align__(16) _Float16 hA[4][16 * DD];   // per-wave A tile (f16)
    __shared__ __align__(16) float    hS[4][16 * DD];   // per-wave h2 tile (f32)

    const int w    = threadIdx.x >> 5;
    const int lane = threadIdx.x & 31;
    const int tile = blockIdx.x * 4 + w;                // 0 .. 4999
    const int b    = tile / (VV / 16);
    const int v0   = (tile % (VV / 16)) * 16;

    // ---- h0 = agg^T + beta*x -> hA (f16) ----
    for (int t = lane; t < 256; t += 32) {              // 256 float4 chunks
        const int m  = t >> 4;
        const int c4 = (t & 15) << 2;
        const float4 a4 = *(const float4*)(agg + (size_t)(v0 + m) * BD + b * DD + c4);
        const float4 x4 = *(const float4*)(x + ((size_t)(b * VV) + v0 + m) * DD + c4);
        const float4 be = *(const float4*)(beta + c4);
        _Float16* p = &hA[w][m * DD + c4];
        p[0] = (_Float16)(a4.x + be.x * x4.x);
        p[1] = (_Float16)(a4.y + be.y * x4.y);
        p[2] = (_Float16)(a4.z + be.z * x4.z);
        p[3] = (_Float16)(a4.w + be.w * x4.w);
    }
    __syncthreads();

    const int hi = (lane >= 16) ? 1 : 0;                // half-wave select
    const int mn = lane & 15;
    v8f acc[4];

    // ---- GEMM1: h1 = relu(h0 @ W1.T + b1) ----
#pragma unroll
    for (int t = 0; t < 4; ++t) {
        const int n = t * 16 + mn;
        const float bias = b1[n];
        v8f c;
#pragma unroll
        for (int r = 0; r < 8; ++r) c[r] = bias;
#pragma unroll
        for (int kk = 0; kk < 2; ++kk) {
            v16h a, bm;
#pragma unroll
            for (int i = 0; i < 16; ++i) {
                // A(16x32 f16): lanes 0-15: K = {0..7, 16..23}; lanes 16-31: {8..15, 24..31}
                const int ka = kk * 32 + (hi ? (i < 8 ? i + 8 : i + 16)
                                             : (i < 8 ? i     : i + 8));
                a[i] = hA[w][mn * DD + ka];
                // B(32x16): same K order as A: lanes 0-15: K = {0..7, 16..23}; lanes 16-31: {8..15, 24..31}
                const int kb = kk * 32 + (hi ? (i < 8 ? i + 8 : i + 16)
                                             : (i < 8 ? i     : i + 8));
                bm[i] = (_Float16)W1[n * DD + kb];
            }
            c = __builtin_amdgcn_wmma_f32_16x16x32_f16(false, a, false, bm,
                                                       (short)0, c, false, false);
        }
        acc[t] = c;
    }
    __syncthreads();

    // ---- ReLU, restage h1 as A (f16) ----
#pragma unroll
    for (int t = 0; t < 4; ++t)
#pragma unroll
        for (int r = 0; r < 8; ++r) {
            const int m = r + hi * 8;
            float v = acc[t][r];
            hA[w][m * DD + t * 16 + mn] = (_Float16)(v > 0.0f ? v : 0.0f);
        }
    __syncthreads();

    // ---- GEMM2: h2 = h1 @ W2.T + b2 ----
#pragma unroll
    for (int t = 0; t < 4; ++t) {
        const int n = t * 16 + mn;
        const float bias = b2[n];
        v8f c;
#pragma unroll
        for (int r = 0; r < 8; ++r) c[r] = bias;
#pragma unroll
        for (int kk = 0; kk < 2; ++kk) {
            v16h a, bm;
#pragma unroll
            for (int i = 0; i < 16; ++i) {
                const int ka = kk * 32 + (hi ? (i < 8 ? i + 8 : i + 16)
                                             : (i < 8 ? i     : i + 8));
                a[i] = hA[w][mn * DD + ka];
                const int kb = kk * 32 + (hi ? (i < 8 ? i + 8 : i + 16)
                                             : (i < 8 ? i     : i + 8));
                bm[i] = (_Float16)W2[n * DD + kb];
            }
            c = __builtin_amdgcn_wmma_f32_16x16x32_f16(false, a, false, bm,
                                                       (short)0, c, false, false);
        }
        acc[t] = c;
    }

    // ---- stage h2 (f32) for LayerNorm ----
#pragma unroll
    for (int t = 0; t < 4; ++t)
#pragma unroll
        for (int r = 0; r < 8; ++r) {
            const int m = r + hi * 8;
            hS[w][m * DD + t * 16 + mn] = acc[t][r];
        }
    __syncthreads();

    // ---- LayerNorm over D + final residual; one lane per row ----
    if (lane < 16) {
        const int m = lane;
        const float* hr = &hS[w][m * DD];
        float mu = 0.0f;
#pragma unroll
        for (int d = 0; d < DD; ++d) mu += hr[d];
        mu *= (1.0f / DD);
        float var = 0.0f;
#pragma unroll
        for (int d = 0; d < DD; ++d) { float dv = hr[d] - mu; var += dv * dv; }
        var *= (1.0f / DD);
        const float inv = rsqrtf(var + LN_EPS);
        const size_t off = ((size_t)(b * VV) + v0 + m) * DD;
#pragma unroll
        for (int d = 0; d < DD; ++d)
            out[off + d] = (hr[d] - mu) * inv * lnw[d] + lnb[d] + x[off + d];
    }
}

// ---------------------------------------------------------------------------
extern "C" void kernel_launch(void* const* d_in, const int* in_sizes, int n_in,
                              void* d_out, int out_size, void* d_ws, size_t ws_size,
                              hipStream_t stream) {
    const float* x    = (const float*)d_in[0];
    const float* z    = (const float*)d_in[1];
    const int*   ei   = (const int*)d_in[2];
    // d_in[3] = r_index (unused by the reference computation)
    const float* Wz   = (const float*)d_in[4];
    const float* bz   = (const float*)d_in[5];
    const float* W1   = (const float*)d_in[6];
    const float* b1   = (const float*)d_in[7];
    const float* W2   = (const float*)d_in[8];
    const float* b2   = (const float*)d_in[9];
    const float* beta = (const float*)d_in[10];
    const float* lnw  = (const float*)d_in[11];
    const float* lnb  = (const float*)d_in[12];
    float* out = (float*)d_out;

    // scratch: relation (16384 f32) + agg (V*256 f32) = ~20.6 MB
    float* rel = (float*)d_ws;
    float* agg = rel + RR * BD;

    k_zero<<<2048, 256, 0, stream>>>(agg, VV * BD);
    k_relation<<<(RR * BD) / 256, 256, 0, stream>>>(z, Wz, bz, rel);
    k_edges<<<2048, 256, 0, stream>>>(ei, x, rel, agg);
    k_mlp_ln<<<(BB * VV / 16) / 4, 128, 0, stream>>>(agg, x, W1, b1, W2, b2,
                                                     beta, lnw, lnb, out);
}
